// CaptioningRNN_33397665693859
// MI455X (gfx1250) — compile-verified, matrix operand order checked by reading, not run on hardware
//
#include <hip/hip_runtime.h>
#include <math.h>

typedef __bf16 bf16;
typedef __attribute__((ext_vector_type(16))) __bf16 v16bf;
typedef __attribute__((ext_vector_type(8)))  __bf16 v8bf;
typedef __attribute__((ext_vector_type(8)))  float  v8f;

#define N_  256
#define T_  64
#define D_  1024
#define H_  1024
#define G4  4096   // 4*H

__device__ __forceinline__ bf16 to_bf16(float f) { return (bf16)f; }

// ---------------------------------------------------------------------------
// Prep kernels (run once per launch)
// ---------------------------------------------------------------------------

// f32 -> bf16 elementwise (for x)
__global__ void cvt_kernel(const float* __restrict__ in, bf16* __restrict__ out, int n) {
  int i = blockIdx.x * blockDim.x + threadIdx.x;
  int stride = gridDim.x * blockDim.x;
  for (; i < n; i += stride) out[i] = to_bf16(in[i]);
}

// W (1024 x 4096, row-major) -> Wt (4096 x 1024, row-major) in bf16.
// After this, B-fragment loads are contiguous-in-K per lane, same as A.
__global__ void wt_kernel(const float* __restrict__ W, bf16* __restrict__ Wt) {
  int idx = blockIdx.x * blockDim.x + threadIdx.x;   // < 4096*1024
  int ncol = idx >> 10;          // 0..4095  (column of W, row of Wt)
  int k    = idx & 1023;         // 0..1023
  Wt[idx] = to_bf16(W[(size_t)k * G4 + ncol]);
}

// h0 = c0 = mean over the 16 spatial keys of A
__global__ void init_h_kernel(const float* __restrict__ Aflat,
                              float* __restrict__ hF, bf16* __restrict__ hB,
                              float* __restrict__ cF) {
  int idx = blockIdx.x * blockDim.x + threadIdx.x;   // < N_*H_
  const float* Ar = Aflat + (size_t)idx * 16;
  float s = 0.f;
#pragma unroll
  for (int k = 0; k < 16; ++k) s += Ar[k];
  s *= (1.f / 16.f);
  hF[idx] = s;
  cF[idx] = s;
  hB[idx] = to_bf16(s);
}

// ---------------------------------------------------------------------------
// Per-step attention: scores -> softmax(16) -> attn, per sample
// ---------------------------------------------------------------------------
#define SRED_S 17   // padded stride: avoids 8-way LDS bank conflicts on 64 banks

__global__ void __launch_bounds__(256)
attn_kernel(const float* __restrict__ Aflat, const float* __restrict__ hF,
            bf16* __restrict__ attnB) {
  __shared__ float sred[256 * SRED_S];
  __shared__ float w[16];
  const int n = blockIdx.x;
  const int tIdx = threadIdx.x;

  float part[16];
#pragma unroll
  for (int k = 0; k < 16; ++k) part[k] = 0.f;

#pragma unroll
  for (int hh = 0; hh < 4; ++hh) {
    int h = tIdx + hh * 256;
    float hv = hF[n * H_ + h];
    const float* Ar = Aflat + ((size_t)n * H_ + h) * 16;
#pragma unroll
    for (int k = 0; k < 16; ++k) part[k] += hv * Ar[k];
  }
#pragma unroll
  for (int k = 0; k < 16; ++k) sred[tIdx * SRED_S + k] = part[k];
  __syncthreads();

  for (int off = 128; off > 0; off >>= 1) {
    if (tIdx < off) {
#pragma unroll
      for (int k = 0; k < 16; ++k)
        sred[tIdx * SRED_S + k] += sred[(tIdx + off) * SRED_S + k];
    }
    __syncthreads();
  }

  if (tIdx == 0) {
    const float scale = 0.03125f;  // 1/sqrt(1024)
    float sc[16];
    float mx = -1e30f;
#pragma unroll
    for (int k = 0; k < 16; ++k) { sc[k] = sred[k] * scale; mx = fmaxf(mx, sc[k]); }
    float sum = 0.f;
#pragma unroll
    for (int k = 0; k < 16; ++k) { sc[k] = __expf(sc[k] - mx); sum += sc[k]; }
    float inv = 1.f / sum;
#pragma unroll
    for (int k = 0; k < 16; ++k) w[k] = sc[k] * inv;
  }
  __syncthreads();

#pragma unroll
  for (int hh = 0; hh < 4; ++hh) {
    int h = tIdx + hh * 256;
    const float* Ar = Aflat + ((size_t)n * H_ + h) * 16;
    float av = 0.f;
#pragma unroll
    for (int k = 0; k < 16; ++k) av += Ar[k] * w[k];
    attnB[n * H_ + h] = to_bf16(av);
  }
}

// ---------------------------------------------------------------------------
// WMMA fragment load: row-major bf16, ISA 16-bit A-matrix layout
//   lanes 0-15 : K = kb+0..7  (VGPR0-3) and kb+16..23 (VGPR4-7)
//   lanes 16-31: K = kb+8..15 (VGPR0-3) and kb+24..31 (VGPR4-7)
// Two global_load_b128 per fragment, contiguous per lane.
// ---------------------------------------------------------------------------
__device__ __forceinline__ v16bf load_frag(const bf16* __restrict__ base, int row,
                                           int stride, int kb, int lane) {
  int koff = kb + ((lane & 16) ? 8 : 0);
  const bf16* p = base + (size_t)row * stride + koff;
  union { v16bf v; v8bf h[2]; } u;
  u.h[0] = *(const v8bf*)(p);
  u.h[1] = *(const v8bf*)(p + 16);
  return u.v;
}

// ---------------------------------------------------------------------------
// Per-step gate GEMM, split-K by source (blockIdx.z):
//   z=0: xt@Wxt   z=1: h@Wht   z=2: attn@Wattnt     (each M=256,N=4096,K=1024)
// Partial f32 results go to aOut + z*N*4H; summed (with bias) in lstm_kernel.
// Block = 256 threads = 8 waves arranged 2(M) x 4(N); wave owns a 32x32 C tile
// = four v_wmma_f32_16x16x32_bf16 accumulators.
// Grid: (4096/128, 256/64, 3) = (32, 4, 3)
// ---------------------------------------------------------------------------
__global__ void __launch_bounds__(256)
gemm_step_kernel(const bf16* __restrict__ xB, int t,
                 const bf16* __restrict__ hB,
                 const bf16* __restrict__ attnB,
                 const bf16* __restrict__ Wxt,
                 const bf16* __restrict__ Wht,
                 const bf16* __restrict__ Wattnt,
                 float* __restrict__ aOut) {
  const int lane = threadIdx.x & 31;
  const int wave = threadIdx.x >> 5;
  const int M0 = blockIdx.y * 64  + (wave >> 2) * 32;
  const int N0 = blockIdx.x * 128 + (wave & 3)  * 32;
  const int mrow = lane & 15;
  const int z = blockIdx.z;

  const bf16* Ab;
  const bf16* Bb;
  int As;
  if (z == 0)      { Ab = xB + (size_t)t * D_; As = T_ * D_; Bb = Wxt;    }
  else if (z == 1) { Ab = hB;                  As = H_;      Bb = Wht;    }
  else             { Ab = attnB;               As = H_;      Bb = Wattnt; }

  v8f acc00 = {}, acc01 = {}, acc10 = {}, acc11 = {};

  for (int kb = 0; kb < 1024; kb += 32) {
    // prefetch next K-slice of the (L2-resident) weight rows
    if (kb + 32 < 1024)
      __builtin_prefetch(Bb + (size_t)(N0 + mrow) * 1024 + kb + 32, 0, 0);

    v16bf a0 = load_frag(Ab, M0 + mrow,      As,   kb, lane);
    v16bf a1 = load_frag(Ab, M0 + 16 + mrow, As,   kb, lane);
    v16bf b0 = load_frag(Bb, N0 + mrow,      1024, kb, lane);
    v16bf b1 = load_frag(Bb, N0 + 16 + mrow, 1024, kb, lane);

    acc00 = __builtin_amdgcn_wmma_f32_16x16x32_bf16(false, a0, false, b0,
                                                    (short)0, acc00, false, false);
    acc01 = __builtin_amdgcn_wmma_f32_16x16x32_bf16(false, a0, false, b1,
                                                    (short)0, acc01, false, false);
    acc10 = __builtin_amdgcn_wmma_f32_16x16x32_bf16(false, a1, false, b0,
                                                    (short)0, acc10, false, false);
    acc11 = __builtin_amdgcn_wmma_f32_16x16x32_bf16(false, a1, false, b1,
                                                    (short)0, acc11, false, false);
  }

  // C layout: VGPR r, lane L -> row = M0 + (L>=16 ? 8:0) + r, col = N0 + (L&15)
  float* __restrict__ po = aOut + (size_t)z * N_ * G4;
  const int col0 = N0 + (lane & 15);
  const int rb   = (lane & 16) ? 8 : 0;
#pragma unroll
  for (int r = 0; r < 8; ++r) {
    int row0 = M0 + rb + r;
    int row1 = M0 + 16 + rb + r;
    po[(size_t)row0 * G4 + col0]      = acc00[r];
    po[(size_t)row0 * G4 + col0 + 16] = acc01[r];
    po[(size_t)row1 * G4 + col0]      = acc10[r];
    po[(size_t)row1 * G4 + col0 + 16] = acc11[r];
  }
}

// ---------------------------------------------------------------------------
// LSTM pointwise update: sums the 3 split-K partials + bias, applies gates
// ---------------------------------------------------------------------------
__global__ void lstm_kernel(const float* __restrict__ aBuf,
                            const float* __restrict__ bias,
                            float* __restrict__ cF, float* __restrict__ hF,
                            bf16* __restrict__ hB,
                            float* __restrict__ out, int t) {
  int idx = blockIdx.x * blockDim.x + threadIdx.x;   // < N_*H_
  int n = idx >> 10;
  int j = idx & 1023;
  const size_t PART = (size_t)N_ * G4;
  const float* a0 = aBuf + (size_t)n * G4;
  const float* a1 = a0 + PART;
  const float* a2 = a1 + PART;

  float ai = a0[j]          + a1[j]          + a2[j]          + bias[j];
  float af = a0[j + H_]     + a1[j + H_]     + a2[j + H_]     + bias[j + H_];
  float ao = a0[j + 2 * H_] + a1[j + 2 * H_] + a2[j + 2 * H_] + bias[j + 2 * H_];
  float ag = a0[j + 3 * H_] + a1[j + 3 * H_] + a2[j + 3 * H_] + bias[j + 3 * H_];

  float ig = 1.f / (1.f + __expf(-ai));
  float fg = 1.f / (1.f + __expf(-af));
  float og = 1.f / (1.f + __expf(-ao));
  float gg = tanhf(ag);
  float c = fg * cF[idx] + ig * gg;
  cF[idx] = c;
  float h = og * tanhf(c);
  hF[idx] = h;
  hB[idx] = to_bf16(h);
  out[((size_t)n * T_ + t) * H_ + j] = h;
}

// ---------------------------------------------------------------------------
// Host-side orchestration
// ---------------------------------------------------------------------------
extern "C" void kernel_launch(void* const* d_in, const int* in_sizes, int n_in,
                              void* d_out, int out_size, void* d_ws, size_t ws_size,
                              hipStream_t stream) {
  const float* x     = (const float*)d_in[0];  // (N,T,D)
  const float* A     = (const float*)d_in[1];  // (N,H,16)
  const float* Wx    = (const float*)d_in[2];  // (D,4H)
  const float* Wh    = (const float*)d_in[3];  // (H,4H)
  const float* Wattn = (const float*)d_in[4];  // (H,4H)
  const float* b     = (const float*)d_in[5];  // (4H,)
  float* out = (float*)d_out;                  // (N,T,H)

  char* ws = (char*)d_ws;
  const size_t WT_BYTES = (size_t)G4 * 1024 * sizeof(bf16);        // 8 MB each
  bf16* Wxt    = (bf16*)ws;                 ws += WT_BYTES;
  bf16* Wht    = (bf16*)ws;                 ws += WT_BYTES;
  bf16* Wattnt = (bf16*)ws;                 ws += WT_BYTES;
  bf16* xB     = (bf16*)ws;                 ws += (size_t)N_ * T_ * D_ * sizeof(bf16); // 32 MB
  bf16* hB     = (bf16*)ws;                 ws += (size_t)N_ * H_ * sizeof(bf16);      // 512 KB
  bf16* attnB  = (bf16*)ws;                 ws += (size_t)N_ * H_ * sizeof(bf16);      // 512 KB
  float* hF    = (float*)ws;                ws += (size_t)N_ * H_ * sizeof(float);     // 1 MB
  float* cF    = (float*)ws;                ws += (size_t)N_ * H_ * sizeof(float);     // 1 MB
  float* aBuf  = (float*)ws;                ws += (size_t)3 * N_ * G4 * sizeof(float); // 12 MB

  // One-time prep
  cvt_kernel<<<4096, 256, 0, stream>>>(x, xB, N_ * T_ * D_);
  wt_kernel<<<16384, 256, 0, stream>>>(Wx, Wxt);
  wt_kernel<<<16384, 256, 0, stream>>>(Wh, Wht);
  wt_kernel<<<16384, 256, 0, stream>>>(Wattn, Wattnt);
  init_h_kernel<<<(N_ * H_) / 256, 256, 0, stream>>>(A, hF, hB, cF);

  // Recurrent loop: attention -> split-K WMMA gate GEMM -> LSTM pointwise
  for (int t = 0; t < T_; ++t) {
    attn_kernel<<<N_, 256, 0, stream>>>(A, hF, attnB);
    gemm_step_kernel<<<dim3(32, 4, 3), 256, 0, stream>>>(xB, t, hB, attnB,
                                                         Wxt, Wht, Wattnt, aBuf);
    lstm_kernel<<<(N_ * H_) / 256, 256, 0, stream>>>(aBuf, b, cF, hF, hB, out, t);
  }
}
